// SwinBasicBlock_22239340658950
// MI455X (gfx1250) — matrix operand order repaired and checked by reading, not compiled, not run on hardware
//
#include <hip/hip_runtime.h>
#include <hip/hip_bf16.h>

// Swin basic block (MI455X / gfx1250, wave32, WMMA f16->f32).
// The reference discards the MLP branch, so only LN -> shift -> window attn
// -> proj -> residual is computed; out = 2*(x_t + attn_residual), NCHW.
//
// Workspace layout (269 MB total):
//   [0,  64MiB)  : y_win f16 (2048 win x 64 tok x 256 C)   -- reused as attn_out
//   [64, 128MiB) : Q  f16 (2048*8 heads x 64 tok x 32 d)
//   [128,192MiB) : K  f16 (same shape)
//   [192,256MiB) : V^T f16 (2048*8 x 32 d x 64 tok)
//   [256MiB ...) : w_qkv^T f16 (768x256), w_proj^T f16 (256x256)

typedef __attribute__((ext_vector_type(16))) _Float16 v16h;
typedef __attribute__((ext_vector_type(8)))  _Float16 v8h;
typedef __attribute__((ext_vector_type(8)))  float    v8f;

#define CC   256
#define HH   128
#define WWID 128
#define NWIN 2048    // 8 batches * 16 * 16 windows
#define NTOK 64

// ---- WMMA fragment helpers (wave32; layouts per CDNA5 ISA) ----

// A fragment: 16x32 f16 tile from a row-major matrix (row stride = ld elems).
// Lane l: row = l&15, half = l>>4.  Element e: K = (e&7) + 8*half + 2*(e&8),
// i.e. two contiguous runs K=[8h,8h+8) and K=[8h+16,8h+24) -> 2x b128 loads.
static __device__ __forceinline__ v16h loadA16x32(const _Float16* base, int ld) {
  int lane = threadIdx.x & 31;
  const _Float16* p = base + (lane & 15) * ld + (lane >> 4) * 8;
  v8h lo = *(const v8h*)(p);
  v8h hi = *(const v8h*)(p + 16);
  v16h a;
  #pragma unroll
  for (int i = 0; i < 8; ++i) { a[i] = lo[i]; a[8 + i] = hi[i]; }
  return a;
}

// B fragment: 32x16 f16 tile, sourced from N-major storage (each row = one
// output column n, K contiguous, row stride ld).  Lane l: n = l&15,
// K = (e&7) + 8*(l>>4) + 2*(e&8)  -> 2x b128 loads (same K order as A).
static __device__ __forceinline__ v16h loadB32x16(const _Float16* bt, int ld) {
  int lane = threadIdx.x & 31;
  const _Float16* p = bt + (lane & 15) * ld + (lane >> 4) * 8;
  v8h lo = *(const v8h*)(p);
  v8h hi = *(const v8h*)(p + 16);
  v16h b;
  #pragma unroll
  for (int i = 0; i < 8; ++i) { b[i] = lo[i]; b[8 + i] = hi[i]; }
  return b;
}

static __device__ __forceinline__ v8f wmma16(v16h a, v16h b, v8f c) {
  return __builtin_amdgcn_wmma_f32_16x16x32_f16(false, a, false, b,
                                                (short)0, c, false, false);
}

// ---- 1) weight transpose + f32->f16 cast ----
__global__ void k_wt(const float* __restrict__ wqkv, const float* __restrict__ wproj,
                     _Float16* __restrict__ wqkvT, _Float16* __restrict__ wprojT) {
  int idx = blockIdx.x * blockDim.x + threadIdx.x;   // 1024*256 = 262144 exactly
  if (idx < 768 * 256) {
    int n = idx >> 8, k = idx & 255;
    wqkvT[n * 256 + k] = (_Float16)wqkv[k * 768 + n];
  } else {
    int i2 = idx - 768 * 256;
    int n = i2 >> 8, k = i2 & 255;
    wprojT[n * 256 + k] = (_Float16)wproj[k * 256 + n];
  }
}

// ---- 2) LayerNorm + roll(-4,-4) + window partition, f32 -> f16 ----
__global__ void k_ln_shift_win(const float* __restrict__ x,
                               const float* __restrict__ g,
                               const float* __restrict__ be,
                               _Float16* __restrict__ ywin) {
  int win = blockIdx.x;               // 2048
  int t   = threadIdx.x;              // 64 tokens per window
  int b   = win >> 8, whi = (win >> 4) & 15, wwi = win & 15;
  int i = t >> 3, j = t & 7;
  int hs = (whi * 8 + i + 4) & 127;   // roll(-4) source row
  int ws = (wwi * 8 + j + 4) & 127;
  const float* px = x + ((size_t)b * CC) * (HH * WWID) + hs * WWID + ws;
  float s = 0.f, ss = 0.f;
  for (int c = 0; c < CC; ++c) {
    float v = px[(size_t)c * (HH * WWID)];
    s += v; ss += v * v;
  }
  float mean = s * (1.f / CC);
  float var  = ss * (1.f / CC) - mean * mean;
  float rstd = rsqrtf(var + 1e-5f);
  _Float16* py = ywin + ((size_t)win * NTOK + t) * CC;
  for (int c = 0; c < CC; ++c) {
    float v = px[(size_t)c * (HH * WWID)];
    py[c] = (_Float16)(((v - mean) * rstd) * g[c] + be[c]);
  }
}

// ---- 3) QKV GEMM: (win*64 x 256) @ (256 x 768), WMMA f16 ----
__global__ void k_qkv(const _Float16* __restrict__ ywin,
                      const _Float16* __restrict__ wqkvT,
                      const float* __restrict__ bqkv,
                      _Float16* __restrict__ Q, _Float16* __restrict__ K,
                      _Float16* __restrict__ VT) {
  int win = blockIdx.x;               // 2048
  int nb  = blockIdx.y;               // 12 blocks of 64 output cols
  int mt  = threadIdx.x >> 5;         // wave -> 16-row M strip
  int lane = threadIdx.x & 31;
  int half = lane >> 4, nl = lane & 15;
  v8f acc[4] = {};
  const _Float16* abase = ywin + ((size_t)win * NTOK + mt * 16) * CC;
  for (int k0 = 0; k0 < CC; k0 += 32) {
    v16h a = loadA16x32(abase + k0, CC);
    #pragma unroll
    for (int nt = 0; nt < 4; ++nt) {
      v16h bf = loadB32x16(wqkvT + (size_t)(nb * 64 + nt * 16) * CC + k0, CC);
      acc[nt] = wmma16(a, bf, acc[nt]);
    }
  }
  #pragma unroll
  for (int nt = 0; nt < 4; ++nt) {
    int n = nb * 64 + nt * 16 + nl;
    float bias = bqkv[n];
    int comp = n >> 8, head = (n >> 5) & 7, d = n & 31;
    size_t bh = (size_t)win * 8 + head;
    #pragma unroll
    for (int v = 0; v < 8; ++v) {
      int tok = mt * 16 + v + 8 * half;        // C/D: row = vgpr + 8*half
      _Float16 val = (_Float16)(acc[nt][v] + bias);
      if (comp == 0)      Q [(bh * 64 + tok) * 32 + d] = val;
      else if (comp == 1) K [(bh * 64 + tok) * 32 + d] = val;
      else                VT[(bh * 32 + d) * 64 + tok] = val;  // store V transposed
    }
  }
}

// ---- 4) windowed attention per (window, head): S=QK^T, softmax, O=PV ----
__global__ void k_attn(const _Float16* __restrict__ Q,
                       const _Float16* __restrict__ K,
                       const _Float16* __restrict__ VT,
                       _Float16* __restrict__ attn) {
  __shared__ float    Ssm[4][16][64];
  __shared__ _Float16 Psm[4][16][64];
  int wh  = blockIdx.x;               // win*8 + head
  int mt  = threadIdx.x >> 5;
  int lane = threadIdx.x & 31;
  int half = lane >> 4, nl = lane & 15;
  const _Float16* qb = Q  + (size_t)wh * 64 * 32;
  const _Float16* kb = K  + (size_t)wh * 64 * 32;
  const _Float16* vb = VT + (size_t)wh * 32 * 64;
  const float scale = 0.17677669529663687f;   // 1/sqrt(32)

  // S strip 16x64: A = Q rows, B = K rows (K storage is already N-major for K^T)
  v16h a = loadA16x32(qb + mt * 16 * 32, 32);
  #pragma unroll
  for (int nt = 0; nt < 4; ++nt) {
    v8f s = {};
    v16h bf = loadB32x16(kb + nt * 16 * 32, 32);
    s = wmma16(a, bf, s);
    #pragma unroll
    for (int v = 0; v < 8; ++v)
      Ssm[mt][v + 8 * half][nt * 16 + nl] = s[v] * scale;
  }
  __syncthreads();

  // softmax: lane pair (half 0/1) splits row nl's 64 cols, combine via shfl_xor 16
  int r = nl, c0 = half * 32;
  float mx = -1e30f;
  for (int jj = 0; jj < 32; ++jj) mx = fmaxf(mx, Ssm[mt][r][c0 + jj]);
  mx = fmaxf(mx, __shfl_xor(mx, 16, 32));
  float sum = 0.f;
  for (int jj = 0; jj < 32; ++jj) {
    float e = __expf(Ssm[mt][r][c0 + jj] - mx);
    Ssm[mt][r][c0 + jj] = e;
    sum += e;
  }
  sum += __shfl_xor(sum, 16, 32);
  float inv = 1.f / sum;
  for (int jj = 0; jj < 32; ++jj)
    Psm[mt][r][c0 + jj] = (_Float16)(Ssm[mt][r][c0 + jj] * inv);
  __syncthreads();

  // O = P(16x64) @ V(64x32); B from V^T (N-major, K contiguous)
  v8f o[2] = {};
  #pragma unroll
  for (int ks = 0; ks < 2; ++ks) {
    v16h ap = loadA16x32(&Psm[mt][0][0] + ks * 32, 64);
    #pragma unroll
    for (int nt = 0; nt < 2; ++nt) {
      v16h bf = loadB32x16(vb + (nt * 16) * 64 + ks * 32, 64);
      o[nt] = wmma16(ap, bf, o[nt]);
    }
  }
  int win = wh >> 3, head = wh & 7;
  #pragma unroll
  for (int nt = 0; nt < 2; ++nt) {
    int d = nt * 16 + nl;
    #pragma unroll
    for (int v = 0; v < 8; ++v) {
      int tok = mt * 16 + v + 8 * half;
      attn[((size_t)win * 64 + tok) * CC + head * 32 + d] = (_Float16)o[nt][v];
    }
  }
}

// ---- 5) proj GEMM + window reverse + roll(+4) + residual + *2 -> NCHW ----
__global__ void k_proj_out(const _Float16* __restrict__ attn,
                           const _Float16* __restrict__ wprojT,
                           const float* __restrict__ bproj,
                           const float* __restrict__ x,
                           float* __restrict__ out) {
  int win = blockIdx.x, nb = blockIdx.y;      // 2048 x 4
  int mt  = threadIdx.x >> 5;
  int lane = threadIdx.x & 31;
  int half = lane >> 4, nl = lane & 15;
  v8f acc[4] = {};
  const _Float16* abase = attn + ((size_t)win * NTOK + mt * 16) * CC;
  for (int k0 = 0; k0 < CC; k0 += 32) {
    v16h a = loadA16x32(abase + k0, CC);
    #pragma unroll
    for (int nt = 0; nt < 4; ++nt) {
      v16h bf = loadB32x16(wprojT + (size_t)(nb * 64 + nt * 16) * CC + k0, CC);
      acc[nt] = wmma16(a, bf, acc[nt]);
    }
  }
  int b = win >> 8, whi = (win >> 4) & 15, wwi = win & 15;
  #pragma unroll
  for (int nt = 0; nt < 4; ++nt) {
    int c = nb * 64 + nt * 16 + nl;
    float bias = bproj[c];
    #pragma unroll
    for (int v = 0; v < 8; ++v) {
      int tok = mt * 16 + v + 8 * half;
      int i = tok >> 3, j = tok & 7;
      int h = (whi * 8 + i + 4) & 127;        // reverse roll (+4)
      int w = (wwi * 8 + j + 4) & 127;
      size_t idx = (((size_t)b * CC + c) * HH + h) * WWID + w;
      out[idx] = 2.f * (x[idx] + (acc[nt][v] + bias));
    }
  }
}

extern "C" void kernel_launch(void* const* d_in, const int* in_sizes, int n_in,
                              void* d_out, int out_size, void* d_ws, size_t ws_size,
                              hipStream_t stream) {
  const float* x     = (const float*)d_in[0];
  const float* n1g   = (const float*)d_in[1];
  const float* n1b   = (const float*)d_in[2];
  const float* wqkv  = (const float*)d_in[3];
  const float* bqkv  = (const float*)d_in[4];
  const float* wproj = (const float*)d_in[5];
  const float* bproj = (const float*)d_in[6];
  // d_in[7..12] (norm2/w1/b1/w2/b2) intentionally unused: MLP output discarded.
  float* out = (float*)d_out;

  char* ws = (char*)d_ws;
  const size_t MB64 = (size_t)64 * 1024 * 1024;
  _Float16* ywin   = (_Float16*)(ws);            // also reused as attn_out
  _Float16* Qb     = (_Float16*)(ws + 1 * MB64);
  _Float16* Kb     = (_Float16*)(ws + 2 * MB64);
  _Float16* VTb    = (_Float16*)(ws + 3 * MB64);
  _Float16* wqkvT  = (_Float16*)(ws + 4 * MB64);
  _Float16* wprojT = (_Float16*)(ws + 4 * MB64 + (size_t)768 * 256 * 2);

  k_wt          <<<1024, 256, 0, stream>>>(wqkv, wproj, wqkvT, wprojT);
  k_ln_shift_win<<<NWIN, 64, 0, stream>>>(x, n1g, n1b, ywin);
  k_qkv         <<<dim3(NWIN, 12), 128, 0, stream>>>(ywin, wqkvT, bqkv, Qb, Kb, VTb);
  k_attn        <<<NWIN * 8, 128, 0, stream>>>(Qb, Kb, VTb, ywin /*attn reuse*/);
  k_proj_out    <<<dim3(NWIN, 4), 128, 0, stream>>>(ywin, wprojT, bproj, x, out);
}
